// GNNEncoderNet_15358803050572
// MI455X (gfx1250) — matrix operand order repaired and checked by reading, not compiled, not run on hardware
//
#include <hip/hip_runtime.h>
#include <hip/hip_bf16.h>

// GNN encoder for MI455X (gfx1250, wave32, WMMA).
// N=100000 nodes, E=1600000 edges, F_IN=128, D=64, OUT=64, ROUNDS=4.

typedef __attribute__((ext_vector_type(16))) __bf16 v16bf;
typedef __attribute__((ext_vector_type(8)))  __bf16 v8bf;
typedef __attribute__((ext_vector_type(2)))  __bf16 v2bf;
typedef __attribute__((ext_vector_type(8)))  float  v8f;

#define GNN_N      100000
#define GNN_FIN    128
#define GNN_D      64
#define GNN_ROUNDS 4

// Weight-fragment table (each 64-wide weight matrix -> 2 or 4 K-blocks x 4 N-blocks,
// each fragment = 32 lanes x 16 bf16 = 1024 bytes, pre-packed in WMMA B layout).
#define FRAG_W1    0    // 16 fragments (K=128)
#define FRAG_W2    16   // 8 fragments
#define FRAG_WREL  24   // 4 rounds x 8
#define FRAG_WROOT 56   // 4 rounds x 8
#define FRAG_WOUT  88   // 8 fragments
#define FRAG_TOTAL 96

// ---- float -> bf16 helpers -------------------------------------------------
// Cheap scalar convert: round-to-nearest (ties away), 2 VALU ops.
__device__ __forceinline__ __bf16 f2bf(float f) {
  union { float f; unsigned u; } c; c.f = f;
  union { unsigned short s; __bf16 b; } r;
  r.s = (unsigned short)((c.u + 0x8000u) >> 16);
  return r.b;
}

// Packed pair convert: single v_cvt_pk_bf16_f32 when available.
__device__ __forceinline__ v2bf f2bf2(float a, float b) {
#if __has_builtin(__builtin_amdgcn_cvt_pk_bf16_f32)
  return __builtin_amdgcn_cvt_pk_bf16_f32(a, b);
#else
  v2bf r; r[0] = f2bf(a); r[1] = f2bf(b); return r;
#endif
}

__device__ __forceinline__ v16bf combine8(v8bf lo, v8bf hi) {
  v16bf a;
#pragma unroll
  for (int j = 0; j < 8; ++j) { a[j] = lo[j]; a[8 + j] = hi[j]; }
  return a;
}

// ---- A-matrix fragment (16x32 bf16) loaders --------------------------------
// ISA 7.12.2: lane L(0-15)=row L, halves 0..7 = K k0..k0+7, halves 8..15 = K k0+16..23;
// lanes 16..31: same rows, K offset +8.
__device__ __forceinline__ v16bf load_A16x32_f32(const float* src, int ld, int row0,
                                                 int k0, int lane) {
  int r  = row0 + (lane & 15);
  int kb = k0 + ((lane & 16) ? 8 : 0);
  const float* p = src + (size_t)r * ld + kb;          // 32B-aligned
  float4 f0 = ((const float4*)p)[0];                   // K+0..3
  float4 f1 = ((const float4*)p)[1];                   // K+4..7
  float4 f2 = ((const float4*)(p + 16))[0];            // K+16..19
  float4 f3 = ((const float4*)(p + 16))[1];            // K+20..23
  v16bf a; v2bf t;
  t = f2bf2(f0.x, f0.y); a[0]  = t[0]; a[1]  = t[1];
  t = f2bf2(f0.z, f0.w); a[2]  = t[0]; a[3]  = t[1];
  t = f2bf2(f1.x, f1.y); a[4]  = t[0]; a[5]  = t[1];
  t = f2bf2(f1.z, f1.w); a[6]  = t[0]; a[7]  = t[1];
  t = f2bf2(f2.x, f2.y); a[8]  = t[0]; a[9]  = t[1];
  t = f2bf2(f2.z, f2.w); a[10] = t[0]; a[11] = t[1];
  t = f2bf2(f3.x, f3.y); a[12] = t[0]; a[13] = t[1];
  t = f2bf2(f3.z, f3.w); a[14] = t[0]; a[15] = t[1];
  return a;
}

// A fragment from a 16x64 bf16 LDS tile (rows 0..15), no conversion needed.
__device__ __forceinline__ v16bf load_A16x32_lds(const __bf16* t, int k0, int lane) {
  int r  = lane & 15;
  int kb = k0 + ((lane & 16) ? 8 : 0);
  const __bf16* p = t + r * GNN_D + kb;                // 16B-aligned
  v8bf lo = *(const v8bf*)p;                           // K+0..7
  v8bf hi = *(const v8bf*)(p + 16);                    // K+16..23
  return combine8(lo, hi);
}

// ---- B-matrix fragment -----------------------------------------------------
// Pre-packed fragment fetch: one aligned 32-byte vector load per lane.
__device__ __forceinline__ v16bf load_Bfrag(const v16bf* __restrict__ frags,
                                            int frag, int lane) {
  return frags[frag * 32 + lane];
}

// One-time strided gather+convert of a 32x16 B fragment from row-major f32 W[K x 64].
// Lane L(0-15) = column col0+L, halves 0..7 = K k0+j, 8..15 = K k0+16..23; lanes 16-31: K offset +8.
__device__ __forceinline__ v16bf cvt_B32x16(const float* __restrict__ W, int k0,
                                            int col0, int lane) {
  int c  = col0 + (lane & 15);
  int kb = k0 + ((lane & 16) ? 8 : 0);
  const float* p = W + (size_t)kb * GNN_D + c;
  v16bf b;
#pragma unroll
  for (int j = 0; j < 16; ++j) b[j] = f2bf(p[(size_t)(j < 8 ? j : j + 8) * GNN_D]);
  return b;
}

__device__ __forceinline__ v8f wmma_bf16(v16bf a, v16bf b, v8f c) {
  // (neg_a, A, neg_b, B, c_mod, C, reuse_a, reuse_b)
  return __builtin_amdgcn_wmma_f32_16x16x32_bf16(false, a, false, b, (short)0, c, false, false);
}

// --- Kernel 0: pack all weights into bf16 B-fragments (96 waves, runs once/launch).
__global__ __launch_bounds__(32) void pack_weights(const float* __restrict__ W1,
                                                   const float* __restrict__ W2,
                                                   const float* __restrict__ Wrel,
                                                   const float* __restrict__ Wroot,
                                                   const float* __restrict__ Wout,
                                                   v16bf* __restrict__ frags) {
  int f    = blockIdx.x;
  int lane = threadIdx.x;
  const float* W; int local;
  if (f < FRAG_W2)         { W = W1;   local = f; }                                  // kb 0..3
  else if (f < FRAG_WREL)  { W = W2;   local = f - FRAG_W2; }
  else if (f < FRAG_WROOT) { int r = (f - FRAG_WREL)  >> 3; W = Wrel  + (size_t)r * GNN_D * GNN_D; local = (f - FRAG_WREL)  & 7; }
  else if (f < FRAG_WOUT)  { int r = (f - FRAG_WROOT) >> 3; W = Wroot + (size_t)r * GNN_D * GNN_D; local = (f - FRAG_WROOT) & 7; }
  else                     { W = Wout; local = f - FRAG_WOUT; }
  int kb = local >> 2;      // K block of 32
  int nb = local & 3;       // N block of 16
  frags[f * 32 + lane] = cvt_B32x16(W, kb * 32, nb * 16, lane);
}

// --- Kernel 1: input MLP.  h = relu(relu(x@W1+b1)@W2+b2).  One wave per 16-row tile.
__global__ __launch_bounds__(32) void mlp_kernel(const float* __restrict__ x,
                                                 const float* __restrict__ b1,
                                                 const float* __restrict__ b2,
                                                 const v16bf* __restrict__ frags,
                                                 float* __restrict__ h) {
  __shared__ __bf16 t1[16 * GNN_D];   // intermediate tile, bf16, row-major
  int lane = threadIdx.x;
  int row0 = blockIdx.x * 16;
  int n16  = lane & 15;
  int mb   = (lane & 16) ? 8 : 0;     // C/D layout: upper half-wave = rows +8

  // stage 1: 16x128 @ 128x64
#pragma unroll
  for (int nb = 0; nb < 4; ++nb) {
    v8f acc = {};
#pragma unroll
    for (int kb = 0; kb < 4; ++kb)
      acc = wmma_bf16(load_A16x32_f32(x, GNN_FIN, row0, kb * 32, lane),
                      load_Bfrag(frags, FRAG_W1 + kb * 4 + nb, lane), acc);
    int n = nb * 16 + n16;
    float bias = b1[n];
#pragma unroll
    for (int v = 0; v < 8; ++v) {
      float val = acc[v] + bias;
      t1[(mb + v) * GNN_D + n] = f2bf(val > 0.f ? val : 0.f);
    }
  }
  __syncthreads();   // single-wave workgroup -> NOP; keeps LDS ordering explicit

  // stage 2: 16x64 @ 64x64 (A from LDS, already bf16)
  v16bf a0 = load_A16x32_lds(t1, 0,  lane);
  v16bf a1 = load_A16x32_lds(t1, 32, lane);
#pragma unroll
  for (int nb = 0; nb < 4; ++nb) {
    v8f acc = {};
    acc = wmma_bf16(a0, load_Bfrag(frags, FRAG_W2 + 0 + nb, lane), acc);
    acc = wmma_bf16(a1, load_Bfrag(frags, FRAG_W2 + 4 + nb, lane), acc);
    int n = nb * 16 + n16;
    float bias = b2[n];
#pragma unroll
    for (int v = 0; v < 8; ++v) {
      float val = acc[v] + bias;
      h[(size_t)(row0 + mb + v) * GNN_D + n] = val > 0.f ? val : 0.f;
    }
  }
}

// --- Kernel 2: edge scatter-add.  agg[dst] += h[src].  One thread per (edge, d).
// 64 contiguous lanes cover one edge's features -> coalesced gather; f32 atomics
// land in the 192MB L2 (whole working set ~140MB is L2-resident).
__global__ __launch_bounds__(256) void scatter_kernel(const float* __restrict__ h,
                                                      const int* __restrict__ ei,
                                                      float* __restrict__ agg,
                                                      int E) {
  long long t = (long long)blockIdx.x * 256 + threadIdx.x;
  int e = (int)(t >> 6);
  int d = (int)(t & 63);
  if (e >= E) return;
  int src = ei[e];
  int dst = ei[E + e];
  float v = h[(size_t)src * GNN_D + d];
  atomicAdd(&agg[(size_t)dst * GNN_D + d], v);
}

// --- Kernel 3: one GraphConv round.  h += relu(agg@Wrel + brel + h@Wroot), in place.
// All A-fragments of h loaded before any store -> no RAW within the tile.
__global__ __launch_bounds__(32) void conv_kernel(float* __restrict__ h,
                                                  const float* __restrict__ agg,
                                                  const v16bf* __restrict__ frags,
                                                  int relBase, int rootBase,
                                                  const float* __restrict__ brel) {
  int lane = threadIdx.x;
  int row0 = blockIdx.x * 16;
  int n16  = lane & 15;
  int mb   = (lane & 16) ? 8 : 0;

  v16bf hA0 = load_A16x32_f32(h,   GNN_D, row0, 0,  lane);
  v16bf hA1 = load_A16x32_f32(h,   GNN_D, row0, 32, lane);
  v16bf gA0 = load_A16x32_f32(agg, GNN_D, row0, 0,  lane);
  v16bf gA1 = load_A16x32_f32(agg, GNN_D, row0, 32, lane);

#pragma unroll
  for (int nb = 0; nb < 4; ++nb) {
    v8f acc = {};
    acc = wmma_bf16(gA0, load_Bfrag(frags, relBase  + 0 + nb, lane), acc);
    acc = wmma_bf16(gA1, load_Bfrag(frags, relBase  + 4 + nb, lane), acc);
    acc = wmma_bf16(hA0, load_Bfrag(frags, rootBase + 0 + nb, lane), acc);
    acc = wmma_bf16(hA1, load_Bfrag(frags, rootBase + 4 + nb, lane), acc);
    int n = nb * 16 + n16;
    float bias = brel[n];
#pragma unroll
    for (int v = 0; v < 8; ++v) {
      size_t idx = (size_t)(row0 + mb + v) * GNN_D + n;
      float c = acc[v] + bias;
      h[idx] = h[idx] + (c > 0.f ? c : 0.f);
    }
  }
}

// --- Kernel 4: output projection.  out = h @ Wout + bout.
__global__ __launch_bounds__(32) void out_kernel(const float* __restrict__ h,
                                                 const v16bf* __restrict__ frags,
                                                 const float* __restrict__ bout,
                                                 float* __restrict__ out) {
  int lane = threadIdx.x;
  int row0 = blockIdx.x * 16;
  int n16  = lane & 15;
  int mb   = (lane & 16) ? 8 : 0;

  v16bf a0 = load_A16x32_f32(h, GNN_D, row0, 0,  lane);
  v16bf a1 = load_A16x32_f32(h, GNN_D, row0, 32, lane);
#pragma unroll
  for (int nb = 0; nb < 4; ++nb) {
    v8f acc = {};
    acc = wmma_bf16(a0, load_Bfrag(frags, FRAG_WOUT + 0 + nb, lane), acc);
    acc = wmma_bf16(a1, load_Bfrag(frags, FRAG_WOUT + 4 + nb, lane), acc);
    int n = nb * 16 + n16;
    float bias = bout[n];
#pragma unroll
    for (int v = 0; v < 8; ++v)
      out[(size_t)(row0 + mb + v) * GNN_D + n] = acc[v] + bias;
  }
}

extern "C" void kernel_launch(void* const* d_in, const int* in_sizes, int n_in,
                              void* d_out, int out_size, void* d_ws, size_t ws_size,
                              hipStream_t stream) {
  const float* x     = (const float*)d_in[0];
  const int*   ei    = (const int*)d_in[1];   // edge_index [2, E]
  // d_in[2] = batch (unused)
  const float* W1    = (const float*)d_in[3];
  const float* b1    = (const float*)d_in[4];
  const float* W2    = (const float*)d_in[5];
  const float* b2    = (const float*)d_in[6];
  const float* Wrel  = (const float*)d_in[7];
  const float* brel  = (const float*)d_in[8];
  const float* Wroot = (const float*)d_in[9];
  const float* Wout  = (const float*)d_in[10];
  const float* bout  = (const float*)d_in[11];

  const int E = in_sizes[1] / 2;

  float* h    = (float*)d_ws;                        // [N, 64] f32 (25.6 MB)
  float* agg  = h + (size_t)GNN_N * GNN_D;           // [N, 64] f32 (25.6 MB)
  v16bf* frags = (v16bf*)(agg + (size_t)GNN_N * GNN_D); // 96 KB bf16 fragments

  const int tiles = GNN_N / 16;                      // 6250, exact
  dim3 waveBlk(32);

  pack_weights<<<FRAG_TOTAL, waveBlk, 0, stream>>>(W1, W2, Wrel, Wroot, Wout, frags);
  mlp_kernel<<<tiles, waveBlk, 0, stream>>>(x, b1, b2, frags, h);

  for (int r = 0; r < GNN_ROUNDS; ++r) {
    hipMemsetAsync(agg, 0, (size_t)GNN_N * GNN_D * sizeof(float), stream);
    long long sthreads = (long long)E * GNN_D;
    int sblocks = (int)((sthreads + 255) / 256);
    scatter_kernel<<<sblocks, 256, 0, stream>>>(h, ei, agg, E);
    conv_kernel<<<tiles, waveBlk, 0, stream>>>(h, agg, frags,
                                               FRAG_WREL  + r * 8,
                                               FRAG_WROOT + r * 8,
                                               brel + (size_t)r * GNN_D);
  }

  out_kernel<<<tiles, waveBlk, 0, stream>>>(h, frags, bout, (float*)d_out);
}
